// MultiHeadAttentionDecomposed_82051055222904
// MI455X (gfx1250) — matrix operand order repaired and checked by reading, not compiled, not run on hardware
//
#include <hip/hip_runtime.h>
#include <hip/hip_bf16.h>

#define D_MODEL   768
#define NUM_HEADS 12
#define D_HEAD    64
#define SEQ       197
#define BATCH     2
#define ROWS      (SEQ * BATCH)   // 394
#define ROWS_PAD  400             // pad to multiple of 16 for WMMA tiles

typedef __attribute__((ext_vector_type(16))) __bf16 v16bf;
typedef __attribute__((ext_vector_type(8)))  float  v8f;

__device__ __forceinline__ unsigned short f32_to_bf16_rne(float f) {
    unsigned u = __float_as_uint(f);
    unsigned r = u + 0x7FFFu + ((u >> 16) & 1u);   // round-to-nearest-even
    return (unsigned short)(r >> 16);
}

// ---------------------------------------------------------------------------
// Kernel 1: convert x -> bf16 rows [r=b*197+m][c] (zero-padded to 400 rows),
//           and W_k -> bf16 (row-major [j][c]).
// ---------------------------------------------------------------------------
__global__ void convert_kernel(const float* __restrict__ x,
                               const float* __restrict__ Wk,
                               unsigned short* __restrict__ Xbf,
                               unsigned short* __restrict__ Wkbf) {
    int idx = blockIdx.x * blockDim.x + threadIdx.x;
    const int NX = ROWS_PAD * D_MODEL;
    if (idx < NX) {
        int r = idx / D_MODEL, c = idx % D_MODEL;
        float v = 0.0f;
        if (r < ROWS) {
            int b = r / SEQ, m = r % SEQ;
            v = x[(m * BATCH + b) * D_MODEL + c];   // x is (SEQ, BATCH, D)
        }
        Xbf[idx] = f32_to_bf16_rne(v);
    } else {
        int j = idx - NX;
        if (j < D_MODEL * D_MODEL) Wkbf[j] = f32_to_bf16_rne(Wk[j]);
    }
}

// ---------------------------------------------------------------------------
// Kernel 2: q_cls[b, j] = (x[n=0,b,:] . Wq[j,:] + Bq[j]) / sqrt(64)
// ---------------------------------------------------------------------------
__global__ void qcls_kernel(const float* __restrict__ x,
                            const float* __restrict__ Wq,
                            const float* __restrict__ Bq,
                            float* __restrict__ qcls) {
    int t = blockIdx.x * blockDim.x + threadIdx.x;
    if (t >= BATCH * D_MODEL) return;
    int b = t / D_MODEL, j = t % D_MODEL;
    const float* xr = x + b * D_MODEL;        // row n=0, batch b
    const float* w  = Wq + j * D_MODEL;
    float acc = Bq[j];
    for (int c = 0; c < D_MODEL; ++c) acc += xr[c] * w[c];
    qcls[t] = acc * 0.125f;                   // 1/sqrt(64)
}

// ---------------------------------------------------------------------------
// Kernel 3: K-projection GEMM via WMMA bf16: Kmat[r, j] = X[r,:] . Wk[j,:] + Bk[j]
// One 16x16 tile per wave, K stepped by 32. 4 waves / block.
// A layout (16-bit 16x32): lane M=lane%16; VGPR g, pair p, half=lane/16:
//   K = half*16 + 2g + p   -> one dword per VGPR (2 consecutive K)
// B layout (16-bit 32x16): lane N=lane%16; K = half*16 + 2g + p
// C/D layout: VGPR v -> row tm*16 + v + 8*half, col tn*16 + lane%16
// ---------------------------------------------------------------------------
__global__ void __launch_bounds__(128)
kproj_wmma_kernel(const unsigned* __restrict__ Xbf,   // bf16 pairs as dwords
                  const unsigned* __restrict__ Wkbf,
                  const float* __restrict__ Bk,
                  float* __restrict__ Kmat) {
    const int wave = threadIdx.x >> 5;
    const int lane = threadIdx.x & 31;
    const int t    = blockIdx.x * 4 + wave;          // tile id, 25*48 = 1200 total
    const int tn   = t % 48;
    const int tm   = t / 48;
    const int M    = lane & 15;
    const int half = lane >> 4;

    const unsigned* arow = Xbf  + (tm * 16 + M) * (D_MODEL / 2);
    const unsigned* brow = Wkbf + (tn * 16 + M) * (D_MODEL / 2);

    v8f acc = {};
    for (int k0 = 0; k0 < D_MODEL; k0 += 32) {
        union { unsigned u[8]; v16bf v; } A, B;
        const int abase = (k0 >> 1) + half * 8;
        const int bbase = (k0 >> 1) + half * 8;
#pragma unroll
        for (int g = 0; g < 8; ++g) {
            A.u[g] = arow[abase + g];
            B.u[g] = brow[bbase + g];
        }
        acc = __builtin_amdgcn_wmma_f32_16x16x32_bf16(
            /*neg_a=*/false, A.v, /*neg_b=*/false, B.v,
            /*c_mod=*/(short)0, acc, /*reuse_a=*/false, /*reuse_b=*/false);
    }

    const float bias = Bk[tn * 16 + M];
#pragma unroll
    for (int v = 0; v < 8; ++v) {
        int row = tm * 16 + v + half * 8;            // < 400 always (padded)
        Kmat[row * D_MODEL + tn * 16 + M] = acc[v] + bias;
    }
}

// ---------------------------------------------------------------------------
// Kernel 4: scores + softmax for the CLS row: one block per (b,h)
// ---------------------------------------------------------------------------
__global__ void attn_softmax_kernel(const float* __restrict__ qcls,
                                    const float* __restrict__ Kmat,
                                    float* __restrict__ attn) {
    const int bh = blockIdx.x;                       // b*12 + h
    const int b  = bh / NUM_HEADS, h = bh % NUM_HEADS;
    const int tid = threadIdx.x;                     // 256 threads
    __shared__ float sred[256];
    __shared__ float sbcast;

    const bool valid = tid < SEQ;
    float sc = 0.0f;
    if (valid) {
        const float* q = qcls + b * D_MODEL + h * D_HEAD;
        const float* k = Kmat + (b * SEQ + tid) * D_MODEL + h * D_HEAD;
        float a = 0.0f;
        for (int d = 0; d < D_HEAD; ++d) a += q[d] * k[d];
        sc = a;
    }
    sred[tid] = valid ? sc : -3.0e38f;
    __syncthreads();
    for (int s = 128; s > 0; s >>= 1) {
        if (tid < s) sred[tid] = fmaxf(sred[tid], sred[tid + s]);
        __syncthreads();
    }
    if (tid == 0) sbcast = sred[0];
    __syncthreads();
    const float mx = sbcast;
    const float e = valid ? __expf(sc - mx) : 0.0f;
    sred[tid] = e;
    __syncthreads();
    for (int s = 128; s > 0; s >>= 1) {
        if (tid < s) sred[tid] += sred[tid + s];
        __syncthreads();
    }
    if (tid == 0) sbcast = sred[0];
    __syncthreads();
    if (valid) attn[bh * SEQ + tid] = e / sbcast;
}

// ---------------------------------------------------------------------------
// Kernel 5: xa[b,h,c] = sum_m attn[b,h,m] * x[b,m,c]  (one block per (b,h))
// ---------------------------------------------------------------------------
__global__ void xa_kernel(const float* __restrict__ attn,
                          const float* __restrict__ x,
                          float* __restrict__ xa) {
    const int bh = blockIdx.x;
    const int b  = bh / NUM_HEADS;
    const int tid = threadIdx.x;                     // 256
    __shared__ float sa[SEQ];
    for (int m = tid; m < SEQ; m += 256) sa[m] = attn[bh * SEQ + m];
    __syncthreads();
    for (int c = tid; c < D_MODEL; c += 256) {
        float acc = 0.0f;
        for (int m = 0; m < SEQ; ++m)
            acc += sa[m] * x[(m * BATCH + b) * D_MODEL + c];
        xa[bh * D_MODEL + c] = acc;
    }
}

// ---------------------------------------------------------------------------
// Kernel 6: u[b,j] = Wv[j,:] . xa[b, h(j), :] + Bv[j]
// ---------------------------------------------------------------------------
__global__ void u_kernel(const float* __restrict__ Wv,
                         const float* __restrict__ Bv,
                         const float* __restrict__ xa,
                         float* __restrict__ u) {
    int t = blockIdx.x * blockDim.x + threadIdx.x;
    if (t >= BATCH * D_MODEL) return;
    int b = t / D_MODEL, j = t % D_MODEL, h = j / D_HEAD;
    const float* w  = Wv + j * D_MODEL;
    const float* xr = xa + (b * NUM_HEADS + h) * D_MODEL;
    float acc = 0.0f;
    for (int c = 0; c < D_MODEL; ++c) acc += w[c] * xr[c];
    u[t] = acc + Bv[j];
}

// ---------------------------------------------------------------------------
// Kernel 7: out[b,o] = Wo[o,:] . u[b,:] + Bo[o]
// ---------------------------------------------------------------------------
__global__ void out_kernel(const float* __restrict__ Wo,
                           const float* __restrict__ Bo,
                           const float* __restrict__ u,
                           float* __restrict__ out) {
    int t = blockIdx.x * blockDim.x + threadIdx.x;
    if (t >= BATCH * D_MODEL) return;
    int b = t / D_MODEL, o = t % D_MODEL;
    const float* w  = Wo + o * D_MODEL;
    const float* ur = u + b * D_MODEL;
    float acc = 0.0f;
    for (int j = 0; j < D_MODEL; ++j) acc += w[j] * ur[j];
    out[b * D_MODEL + o] = acc + Bo[o];
}

// ---------------------------------------------------------------------------
// Host launcher
// ---------------------------------------------------------------------------
extern "C" void kernel_launch(void* const* d_in, const int* in_sizes, int n_in,
                              void* d_out, int out_size, void* d_ws, size_t ws_size,
                              hipStream_t stream) {
    const float* x  = (const float*)d_in[0];
    const float* Wq = (const float*)d_in[1];
    const float* Wk = (const float*)d_in[2];
    const float* Wv = (const float*)d_in[3];
    const float* Wo = (const float*)d_in[4];
    const float* Bq = (const float*)d_in[5];
    const float* Bk = (const float*)d_in[6];
    const float* Bv = (const float*)d_in[7];
    const float* Bo = (const float*)d_in[8];
    float* out = (float*)d_out;

    // workspace layout (bytes), ~3.13 MB total
    char* ws = (char*)d_ws;
    unsigned short* Xbf  = (unsigned short*)(ws + 0);                 // 400*768*2  = 614400
    unsigned short* Wkbf = (unsigned short*)(ws + 614400);            // 768*768*2  = 1179648
    float* Kmat = (float*)(ws + 1794048);                             // 400*768*4  = 1228800
    float* qcls = (float*)(ws + 3022848);                             // 2*768*4    = 6144
    float* attn = (float*)(ws + 3028992);                             // 24*197*4   = 18912
    float* xa   = (float*)(ws + 3047936);                             // 24*768*4   = 73728
    float* u    = (float*)(ws + 3121664);                             // 2*768*4    = 6144

    // 1. bf16 conversion (x padded to 400 rows, Wk)
    {
        int total = ROWS_PAD * D_MODEL + D_MODEL * D_MODEL;           // 897024
        convert_kernel<<<(total + 255) / 256, 256, 0, stream>>>(x, Wk, Xbf, Wkbf);
    }
    // 2. q_cls
    qcls_kernel<<<(BATCH * D_MODEL + 255) / 256, 256, 0, stream>>>(x, Wq, Bq, qcls);
    // 3. K projection via WMMA: 25*48 = 1200 tiles, 4 waves/block -> 300 blocks
    kproj_wmma_kernel<<<300, 128, 0, stream>>>((const unsigned*)Xbf, (const unsigned*)Wkbf, Bk, Kmat);
    // 4. scores + softmax (CLS row only)
    attn_softmax_kernel<<<BATCH * NUM_HEADS, 256, 0, stream>>>(qcls, Kmat, attn);
    // 5. attention-weighted input sum
    xa_kernel<<<BATCH * NUM_HEADS, 256, 0, stream>>>(attn, x, xa);
    // 6. Wv contraction (+Bv)
    u_kernel<<<(BATCH * D_MODEL + 255) / 256, 256, 0, stream>>>(Wv, Bv, xa, u);
    // 7. output projection (+Bo)
    out_kernel<<<(BATCH * D_MODEL + 255) / 256, 256, 0, stream>>>(Wo, Bo, u, out);
}
